// CoAttention_395136991716
// MI455X (gfx1250) — matrix operand order repaired and checked by reading, not compiled, not run on hardware
//
#include <hip/hip_runtime.h>

// Problem constants (from the reference)
#define QD   768   // Q_DIM
#define VD   512   // V_DIM
#define HIDD 512   // HID
#define NB   64    // batch
#define NQ   512
#define NV   1024

// LDS tile row pitch in bf16 elements: 64 + 8 pad -> 144B stride.
// Bank base per row = 36*row mod 64: distinct for 16 consecutive rows.
#define PADK 72

typedef __attribute__((ext_vector_type(16))) __bf16 v16bf;
typedef __attribute__((ext_vector_type(8)))  float  v8f;
typedef __attribute__((__vector_size__(4 * sizeof(int)))) int v4i;

union FragBF { uint4 u[2]; v16bf v; };

__device__ __forceinline__ v8f wmma_bf16(v16bf a, v16bf b, v8f c) {
    // (neg_a, A, neg_b, B, c_mod, C, reuse_a, reuse_b)
    return __builtin_amdgcn_wmma_f32_16x16x32_bf16(false, a, false, b, (short)0, c,
                                                   false, false);
}

// --- CDNA5 async global->LDS path (guarded; fallback = register staging) ----
#if __has_builtin(__builtin_amdgcn_global_load_async_to_lds_b128)
#define HAVE_ASYNC_LDS 1
typedef __attribute__((address_space(1))) v4i gas_v4i;   // global AS pointer elem
typedef __attribute__((address_space(3))) v4i las_v4i;   // LDS AS pointer elem
#else
#define HAVE_ASYNC_LDS 0
#endif

#if __has_builtin(__builtin_amdgcn_s_wait_asynccnt)
#define WAIT_ASYNC(n) __builtin_amdgcn_s_wait_asynccnt(n)
#else
#define WAIT_ASYNC(n) asm volatile("s_wait_asynccnt %0" ::"n"(n) : "memory")
#endif

// Stage a 64x64 bf16 tile (row stride HIDD in global) into LDS (pitch PADK).
// 128 threads x 4 chunks x 16B == 64 rows x 128B.
__device__ __forceinline__ void stage_tile(const __bf16* __restrict__ g0,
                                           __bf16* l0, int tid) {
#pragma unroll
    for (int i = 0; i < 4; ++i) {
        const int chunk = tid + 128 * i;
        const int row   = chunk >> 3;
        const int c16   = chunk & 7;                 // 16B chunk within row
        const __bf16* g = g0 + (size_t)row * HIDD + c16 * 8;
        __bf16*       l = l0 + row * PADK + c16 * 8;
#if HAVE_ASYNC_LDS
        __builtin_amdgcn_global_load_async_to_lds_b128(
            (gas_v4i*)(void*)g, (las_v4i*)(void*)l, 0, 0);
#else
        *reinterpret_cast<uint4*>(l) = *reinterpret_cast<const uint4*>(g);
#endif
    }
}

// ---------------------------------------------------------------------------
// Kernel 1: out[row,n] = sum_d X[row,d]*W[n,d] + bias[n]   (fp32 in, bf16 out)
// Block: 128 threads = 4 waves. Each wave: 16 rows x 64 cols (4 WMMA C tiles).
// A-frag layout (16x32 bf16): lane L -> row m0+(L&15); halves j<8: K=kb+j,
// j>=8: K=kb+16+(j-8), kb = (L<16?0:8).  B-frag (32x16): lane L -> col
// n0+(L&15); same K order as A: j<8: K=kb+j, j>=8: K=kb+16+(j-8).
// ---------------------------------------------------------------------------
__global__ __launch_bounds__(128)
void proj_kernel(const float* __restrict__ X, const float* __restrict__ W,
                 const float* __restrict__ bias, __bf16* __restrict__ out, int D) {
    const int lane   = threadIdx.x & 31;
    const int wave   = threadIdx.x >> 5;
    const int l15    = lane & 15;
    const int half16 = lane >> 4;           // 0 or 1
    const int m0     = blockIdx.x * 64 + wave * 16;
    const int n0     = blockIdx.y * 64;

    const float* xrow = X + (size_t)(m0 + l15) * D;
    v8f acc[4] = {};

    for (int kk = 0; kk < D; kk += 32) {
        FragBF a;
        {
            const float* ap = xrow + kk + half16 * 8;
            __builtin_prefetch(ap + 32, 0, 0);      // next k-step
            const float4 c0 = *reinterpret_cast<const float4*>(ap);
            const float4 c1 = *reinterpret_cast<const float4*>(ap + 4);
            const float4 c2 = *reinterpret_cast<const float4*>(ap + 16);
            const float4 c3 = *reinterpret_cast<const float4*>(ap + 20);
            a.v[0]=(__bf16)c0.x; a.v[1]=(__bf16)c0.y; a.v[2]=(__bf16)c0.z; a.v[3]=(__bf16)c0.w;
            a.v[4]=(__bf16)c1.x; a.v[5]=(__bf16)c1.y; a.v[6]=(__bf16)c1.z; a.v[7]=(__bf16)c1.w;
            a.v[8]=(__bf16)c2.x; a.v[9]=(__bf16)c2.y; a.v[10]=(__bf16)c2.z; a.v[11]=(__bf16)c2.w;
            a.v[12]=(__bf16)c3.x; a.v[13]=(__bf16)c3.y; a.v[14]=(__bf16)c3.z; a.v[15]=(__bf16)c3.w;
        }
#pragma unroll
        for (int nt = 0; nt < 4; ++nt) {
            const int col   = n0 + nt * 16 + l15;
            const float* wr = W + (size_t)col * D + kk + half16 * 8;
            FragBF b;
#pragma unroll
            for (int j = 0; j < 16; j += 4) {
                const float4 f = *reinterpret_cast<const float4*>(wr + j + (j & 8));
                b.v[j+0]=(__bf16)f.x; b.v[j+1]=(__bf16)f.y;
                b.v[j+2]=(__bf16)f.z; b.v[j+3]=(__bf16)f.w;
            }
            acc[nt] = wmma_bf16(a.v, b.v, acc[nt]);
        }
    }

    // Epilogue: +bias, convert to bf16, store. C layout: lane holds col, rows
    // m0 + half16*8 + r for VGPR r.
#pragma unroll
    for (int nt = 0; nt < 4; ++nt) {
        const int col  = n0 + nt * 16 + l15;
        const float bv = bias[col];
#pragma unroll
        for (int r = 0; r < 8; ++r) {
            const int row = m0 + half16 * 8 + r;
            out[(size_t)row * HIDD + col] = (__bf16)(acc[nt][r] + bv);
        }
    }
}

// ---------------------------------------------------------------------------
// Kernel 2: per batch b, E[q,v] = exp(tanh( Qp[b]·Vp[b]^T )) (K = HID = 512).
// Async-LDS double-buffered GEMM: both 64x64 bf16 tiles staged cooperatively
// via GLOBAL_LOAD_ASYNC_TO_LDS_B128 (ASYNCcnt), fragments read with
// ds_load_b128 (conflict-free, padded pitch). tanh in [-1,1] => softmax
// needs no max subtraction; stores E (fp16) + row sums R, col sums C.
// ---------------------------------------------------------------------------
__global__ __launch_bounds__(128)
void hmap_kernel(const __bf16* __restrict__ Qp, const __bf16* __restrict__ Vp,
                 _Float16* __restrict__ E, float* __restrict__ R,
                 float* __restrict__ C) {
    __shared__ __bf16 As[2][64 * PADK];
    __shared__ __bf16 Bs[2][64 * PADK];

    const int b      = blockIdx.z;
    const int tid    = threadIdx.x;
    const int lane   = tid & 31;
    const int wave   = tid >> 5;
    const int l15    = lane & 15;
    const int half16 = lane >> 4;
    const int q0     = blockIdx.x * 64;          // block's q tile
    const int v0     = blockIdx.y * 64;          // block's v tile

    const __bf16* qbase = Qp + ((size_t)b * NQ + q0) * HIDD;
    const __bf16* vbase = Vp + ((size_t)b * NV + v0) * HIDD;

    // Prologue: stage K-stage 0 into buffer 0 (8 async b128 per thread).
    stage_tile(qbase, As[0], tid);
    stage_tile(vbase, Bs[0], tid);

    v8f acc[4] = {};
    int s = 0;
    for (int kk = 0; kk < HIDD; kk += 64, s ^= 1) {
        const bool more = (kk + 64) < HIDD;
        if (more) {                               // prefetch next stage
            stage_tile(qbase + kk + 64, As[s ^ 1], tid);
            stage_tile(vbase + kk + 64, Bs[s ^ 1], tid);
        }
#if HAVE_ASYNC_LDS
        if (more) WAIT_ASYNC(8);                  // fence current stage only
        else      WAIT_ASYNC(0);
#endif
        __syncthreads();

#pragma unroll
        for (int ks = 0; ks < 64; ks += 32) {
            FragBF a;
            const __bf16* ap = &As[s][(wave * 16 + l15) * PADK + ks + half16 * 8];
            a.u[0] = *reinterpret_cast<const uint4*>(ap);
            a.u[1] = *reinterpret_cast<const uint4*>(ap + 16);
#pragma unroll
            for (int nt = 0; nt < 4; ++nt) {
                const __bf16* bp = &Bs[s][(nt * 16 + l15) * PADK + ks + half16 * 8];
                FragBF bb;
                bb.u[0] = *reinterpret_cast<const uint4*>(bp);
                bb.u[1] = *reinterpret_cast<const uint4*>(bp + 16);
                acc[nt] = wmma_bf16(a.v, bb.v, acc[nt]);
            }
        }
        __syncthreads();                          // buffer reuse fence
    }

    // Epilogue: e = exp(tanh(h)); store E fp16; accumulate col/row sums.
    float rsum[8] = {0.f,0.f,0.f,0.f,0.f,0.f,0.f,0.f};
#pragma unroll
    for (int nt = 0; nt < 4; ++nt) {
        const int col = v0 + nt * 16 + l15;
        float csum = 0.f;
#pragma unroll
        for (int r = 0; r < 8; ++r) {
            const float e = __expf(tanhf(acc[nt][r]));
            const int q   = q0 + wave * 16 + half16 * 8 + r;
            E[((size_t)b * NQ + q) * NV + col] = (_Float16)e;
            csum    += e;
            rsum[r] += e;
        }
        atomicAdd(&C[b * NV + col], csum);   // two lanes (L, L+16) share a col
    }
    // Row sums: reduce across the 16 lanes holding the 16 cols of this tile
    // (xor offsets 1..8 stay within each 16-lane half; halves hold different rows).
#pragma unroll
    for (int off = 1; off < 16; off <<= 1)
#pragma unroll
        for (int r = 0; r < 8; ++r)
            rsum[r] += __shfl_xor(rsum[r], off, 32);
    if (l15 == 0) {
#pragma unroll
        for (int r = 0; r < 8; ++r)
            atomicAdd(&R[b * NQ + q0 + wave * 16 + half16 * 8 + r], rsum[r]);
    }
}

// ---------------------------------------------------------------------------
// Kernel 3: w_v[b,v] = sum_q E[q,v]/R[q] ;  w_q[b,q] = sum_v E[q,v]/C[v].
// Block handles 64 q-rows x all 1024 v (coalesced E reads). Grid (NQ/64, B).
// ---------------------------------------------------------------------------
__global__ __launch_bounds__(256)
void wsum_kernel(const _Float16* __restrict__ E, const float* __restrict__ R,
                 const float* __restrict__ C, float* __restrict__ wv,
                 float* __restrict__ wq) {
    __shared__ float rowacc[64];
    const int b  = blockIdx.y;
    const int q0 = blockIdx.x * 64;
    const int t  = threadIdx.x;
    if (t < 64) rowacc[t] = 0.f;
    __syncthreads();

    float wvloc[4] = {0.f,0.f,0.f,0.f};
    float cinv[4];
#pragma unroll
    for (int i = 0; i < 4; ++i)
        cinv[i] = 1.0f / C[b * NV + t + 256 * i];

    for (int qi = 0; qi < 64; ++qi) {
        const int q      = q0 + qi;
        const float rinv = 1.0f / R[b * NQ + q];
        const _Float16* erow = E + ((size_t)b * NQ + q) * NV;
        float rloc = 0.f;
#pragma unroll
        for (int i = 0; i < 4; ++i) {
            const float e = (float)erow[t + 256 * i];
            wvloc[i] += e * rinv;
            rloc     += e * cinv[i];
        }
        // wave reduction, then one LDS atomic per wave
#pragma unroll
        for (int off = 16; off >= 1; off >>= 1)
            rloc += __shfl_xor(rloc, off, 32);
        if ((t & 31) == 0) atomicAdd(&rowacc[qi], rloc);
    }
    __syncthreads();
    if (t < 64) wq[b * NQ + q0 + t] = rowacc[t];   // unique block per (b,q)
#pragma unroll
    for (int i = 0; i < 4; ++i)
        atomicAdd(&wv[b * NV + t + 256 * i], wvloc[i]);  // 8 blocks contribute
}

// ---------------------------------------------------------------------------
// Kernel 4: out[b,d] = sum_n w[b,n] * X[b,n,d]   (memory-bound matvec)
// ---------------------------------------------------------------------------
__global__ __launch_bounds__(256)
void outmv_kernel(const float* __restrict__ w, const float* __restrict__ X,
                  float* __restrict__ out, int N, int D) {
    const int b = blockIdx.y;
    const int d = blockIdx.x * 256 + threadIdx.x;
    const float* xb = X + (size_t)b * N * D;
    const float* wb = w + (size_t)b * N;
    float acc = 0.f;
    for (int n = 0; n < N; ++n)
        acc = fmaf(wb[n], xb[(size_t)n * D + d], acc);
    out[(size_t)b * D + d] = acc;
}

// ---------------------------------------------------------------------------
extern "C" void kernel_launch(void* const* d_in, const int* in_sizes, int n_in,
                              void* d_out, int out_size, void* d_ws, size_t ws_size,
                              hipStream_t stream) {
    const float* Vf   = (const float*)d_in[0];   // [64,1024,512]
    const float* Qf   = (const float*)d_in[1];   // [64,512,768]
    const float* Wq_w = (const float*)d_in[2];   // [512,768]
    const float* Wq_b = (const float*)d_in[3];   // [512]
    const float* Wv_w = (const float*)d_in[4];   // [512,512]
    const float* Wv_b = (const float*)d_in[5];   // [512]
    float* out = (float*)d_out;                  // v_hat [64,512] ++ q_hat [64,768]

    // Workspace layout (bytes)
    char* ws = (char*)d_ws;
    size_t off = 0;
    __bf16*    Qp = (__bf16*)(ws + off);    off += (size_t)NB * NQ * HIDD * 2;  // 33.5 MB
    __bf16*    Vp = (__bf16*)(ws + off);    off += (size_t)NB * NV * HIDD * 2;  // 67 MB
    _Float16*  E  = (_Float16*)(ws + off);  off += (size_t)NB * NQ * NV * 2;    // 67 MB
    float*     R  = (float*)(ws + off);     off += (size_t)NB * NQ * 4;
    float*     C  = (float*)(ws + off);     off += (size_t)NB * NV * 4;
    float*     wv = (float*)(ws + off);     off += (size_t)NB * NV * 4;
    float*     wq = (float*)(ws + off);     off += (size_t)NB * NQ * 4;

    // Zero the accumulators (R, C, wv, wq are contiguous) — capture-safe.
    (void)hipMemsetAsync(R, 0, (size_t)(NB * NQ + NB * NV + NB * NV + NB * NQ) * 4, stream);

    // 1) Projections (fp32 -> bf16 WMMA -> bf16 activations)
    proj_kernel<<<dim3((NB * NQ) / 64, HIDD / 64), 128, 0, stream>>>(Qf, Wq_w, Wq_b, Qp, QD);
    proj_kernel<<<dim3((NB * NV) / 64, HIDD / 64), 128, 0, stream>>>(Vf, Wv_w, Wv_b, Vp, VD);

    // 2) E = exp(tanh(Qp·Vp^T)) + row/col sums (async-LDS double-buffered WMMA)
    hmap_kernel<<<dim3(NQ / 64, NV / 64, NB), 128, 0, stream>>>(Qp, Vp, E, R, C);

    // 3) Softmax column-sum weights
    wsum_kernel<<<dim3(NQ / 64, NB), 256, 0, stream>>>(E, R, C, wv, wq);

    // 4) v_hat = wv·V ; q_hat = wq·Q
    outmv_kernel<<<dim3(VD / 256, NB), 256, 0, stream>>>(wv, Vf, out, NV, VD);
    outmv_kernel<<<dim3(QD / 256, NB), 256, 0, stream>>>(wq, Qf, out + (size_t)NB * VD, NQ, QD);

    (void)in_sizes; (void)n_in; (void)out_size; (void)ws_size;
}
